// HadamardAttention_4277787427018
// MI455X (gfx1250) — matrix operand order repaired and checked by reading, not compiled, not run on hardware
//
#include <hip/hip_runtime.h>
#include <stdint.h>

// ---------------------------------------------------------------------------
// Problem constants (from reference): B=8, N=4096, C=1024, H=32, D=32
// ---------------------------------------------------------------------------
#define BB    8
#define NN    4096
#define CC    1024
#define HH    32
#define DD    32
#define MTOK  (BB * NN)          // 32768 tokens
#define KT    (CC / 32)          // 32 K-tiles of 32
#define NT    (CC / 16)          // 64 N-tiles of 16
#define SCALE 0.17677669529663687f   // 32^-0.5

typedef __attribute__((ext_vector_type(16))) __bf16 v16bf;
typedef __attribute__((ext_vector_type(8)))  float  v8f;

union Frag16 { v16bf v; uint4 q[2]; };
union Pack8  { unsigned short h[8];  uint4 q;    };
union Pack16 { unsigned short h[16]; uint4 q[2]; };

__device__ __forceinline__ unsigned short f2bf(float f) {
  uint32_t u = __float_as_uint(f);
  u += 0x7FFFu + ((u >> 16) & 1u);          // round-to-nearest-even
  return (unsigned short)(u >> 16);
}
__device__ __forceinline__ float bf2f(unsigned short h) {
  return __uint_as_float((uint32_t)h << 16);
}

// ---------------------------------------------------------------------------
// K0: fp32 -> bf16 conversion of x (8 elements / thread, b128 in & out)
// ---------------------------------------------------------------------------
__global__ void cvt_bf16_kernel(const float* __restrict__ x,
                                unsigned short* __restrict__ xb, int n8) {
  int i = blockIdx.x * 256 + threadIdx.x;
  if (i >= n8) return;
  const float4* s = reinterpret_cast<const float4*>(x) + (size_t)i * 2;
  float4 a = s[0], b = s[1];
  Pack8 u;
  u.h[0] = f2bf(a.x); u.h[1] = f2bf(a.y); u.h[2] = f2bf(a.z); u.h[3] = f2bf(a.w);
  u.h[4] = f2bf(b.x); u.h[5] = f2bf(b.y); u.h[6] = f2bf(b.z); u.h[7] = f2bf(b.w);
  reinterpret_cast<uint4*>(xb)[i] = u.q;
}

// ---------------------------------------------------------------------------
// K1: pack a 1024x1024 fp32 weight into bf16 WMMA B-fragment layout.
// Fragment for tile (kt,nt): lane holds col = nt*16+(lane&15),
// k = kt*32 + (lane>>4)*16 + e  (e = 0..15), stored contiguously.
// ---------------------------------------------------------------------------
__global__ void pack_w_kernel(const float* __restrict__ W,
                              unsigned short* __restrict__ out) {
  int tid  = blockIdx.x * 256 + threadIdx.x;       // 65536 threads
  int lane = tid & 31;
  int tile = tid >> 5;                              // 0..2047
  int kt   = tile & (KT - 1);
  int nt   = tile >> 5;                             // tile / KT
  int col  = nt * 16 + (lane & 15);
  int kbase = kt * 32 + (lane >> 4) * 16;
  Pack16 t;
#pragma unroll
  for (int e = 0; e < 16; e++)
    t.h[e] = f2bf(W[(size_t)(kbase + e) * CC + col]);
  uint4* dst = reinterpret_cast<uint4*>(out) + (size_t)(tile * 32 + lane) * 2;
  dst[0] = t.q[0];
  dst[1] = t.q[1];
}

// ---------------------------------------------------------------------------
// K2: fused QKV projection + per-token Hadamard dot.
// One block = 16-token panel. A panel staged in LDS; 8 waves x 8 N-tiles;
// 3 WMMAs per K-step sharing the A fragment. Emits v (bf16) and
// aw_raw[token][h] = SCALE * sum_d q*k (biases included exactly).
// ---------------------------------------------------------------------------
__global__ __launch_bounds__(256) void qkv_dot_kernel(
    const unsigned short* __restrict__ xb,
    const unsigned short* __restrict__ pWq,
    const unsigned short* __restrict__ pWk,
    const unsigned short* __restrict__ pWv,
    const float* __restrict__ bq, const float* __restrict__ bk,
    const float* __restrict__ bv,
    unsigned short* __restrict__ v_out, float* __restrict__ aw_raw) {
  __shared__ unsigned short smA[16 * CC];     // 32 KB A panel (bf16)
  __shared__ float red[16][NT];               // per-(row, tile) q.k partials

  const int tid = threadIdx.x;
  const int mb  = blockIdx.x;                 // token panel index

  {   // stage contiguous 16x1024 bf16 panel
    const uint4* src = reinterpret_cast<const uint4*>(xb + (size_t)mb * 16 * CC);
    uint4* dst = reinterpret_cast<uint4*>(smA);
#pragma unroll
    for (int i = 0; i < 8; i++) dst[tid + i * 256] = src[tid + i * 256];
  }
  __syncthreads();

  const int wave = tid >> 5, lane = tid & 31, half = lane >> 4, mrow = lane & 15;
  const uint4* q4 = reinterpret_cast<const uint4*>(pWq);
  const uint4* k4 = reinterpret_cast<const uint4*>(pWk);
  const uint4* v4 = reinterpret_cast<const uint4*>(pWv);

  for (int t = wave; t < NT; t += 8) {
    v8f qa, ka, va;
#pragma unroll
    for (int i = 0; i < 8; i++) { qa[i] = 0.f; ka[i] = 0.f; va[i] = 0.f; }

    for (int kb = 0; kb < KT; kb++) {
      Frag16 A, Bq, Bk, Bv;
      // A fragment (contiguous k order, same as packed B): lanes0-15 K in {0..15}, lanes16-31 K in {16..31}
      int abase = mrow * CC + kb * 32 + half * 16;
      A.q[0] = *reinterpret_cast<const uint4*>(&smA[abase]);
      A.q[1] = *reinterpret_cast<const uint4*>(&smA[abase + 8]);
      size_t wb = ((size_t)(t * KT + kb) * 32 + lane) * 2;
      Bq.q[0] = q4[wb]; Bq.q[1] = q4[wb + 1];
      Bk.q[0] = k4[wb]; Bk.q[1] = k4[wb + 1];
      Bv.q[0] = v4[wb]; Bv.q[1] = v4[wb + 1];
      __builtin_prefetch((const void*)(q4 + wb + 64), 0, 1);
      __builtin_prefetch((const void*)(v4 + wb + 64), 0, 1);
      qa = __builtin_amdgcn_wmma_f32_16x16x32_bf16(false, A.v, false, Bq.v,
                                                   (short)0, qa, false, false);
      ka = __builtin_amdgcn_wmma_f32_16x16x32_bf16(false, A.v, false, Bk.v,
                                                   (short)0, ka, false, false);
      va = __builtin_amdgcn_wmma_f32_16x16x32_bf16(false, A.v, false, Bv.v,
                                                   (short)0, va, false, false);
    }

    // epilogue: bias, v store, Hadamard partial reduce
    int   col = t * 16 + mrow;
    float bqs = bq[col], bks = bk[col], bvs = bv[col];
    float s[8];
#pragma unroll
    for (int vi = 0; vi < 8; vi++) {
      float qv = qa[vi] + bqs, kv = ka[vi] + bks, vv = va[vi] + bvs;
      int row = mb * 16 + vi + half * 8;          // D layout: lanes16-31 -> M+8
      v_out[(size_t)row * CC + col] = f2bf(vv);
      s[vi] = qv * kv;
    }
#pragma unroll
    for (int vi = 0; vi < 8; vi++) {              // reduce over 16-lane half
      s[vi] += __shfl_xor(s[vi], 8);
      s[vi] += __shfl_xor(s[vi], 4);
      s[vi] += __shfl_xor(s[vi], 2);
      s[vi] += __shfl_xor(s[vi], 1);
    }
    if (mrow == 0) {
#pragma unroll
      for (int vi = 0; vi < 8; vi++) red[vi + half * 8][t] = s[vi];
    }
  }
  __syncthreads();

  for (int idx = tid; idx < 16 * HH; idx += 256) {
    int row = idx >> 5, h = idx & 31;
    int token = mb * 16 + row;
    aw_raw[(size_t)token * HH + h] = (red[row][2 * h] + red[row][2 * h + 1]) * SCALE;
  }
}

// ---------------------------------------------------------------------------
// K3: per-token dak MLP: relu(aw@W1+b1)@W2+b2, then mask
// ---------------------------------------------------------------------------
__global__ void dak_mlp_kernel(const float* __restrict__ aw_raw,
                               const int* __restrict__ mask,
                               const float* __restrict__ W1,
                               const float* __restrict__ b1,
                               const float* __restrict__ W2,
                               const float* __restrict__ b2,
                               float* __restrict__ aw_mlp) {
  int token = blockIdx.x * 256 + threadIdx.x;
  if (token >= MTOK) return;
  float ain[HH], o[DD];
#pragma unroll
  for (int i = 0; i < HH; i++) ain[i] = aw_raw[(size_t)token * HH + i];
#pragma unroll
  for (int d = 0; d < DD; d++) o[d] = b2[d];
  for (int j = 0; j < 2 * DD; j++) {
    float s = b1[j];
#pragma unroll
    for (int i = 0; i < HH; i++) s += ain[i] * W1[i * (2 * DD) + j];
    s = fmaxf(s, 0.f);
#pragma unroll
    for (int d = 0; d < DD; d++) o[d] += s * W2[j * DD + d];
  }
  bool dead = (mask[token] == 0);
#pragma unroll
  for (int d = 0; d < DD; d++)
    aw_mlp[(size_t)token * HH + d] = dead ? -1e9f : o[d];
}

// ---------------------------------------------------------------------------
// K4: softmax over sequence dim N per (b, h). One block per (b,h).
// ---------------------------------------------------------------------------
__global__ void softmax_n_kernel(const float* __restrict__ in,
                                 float* __restrict__ out) {
  int b = blockIdx.x >> 5, h = blockIdx.x & 31;
  size_t base = (size_t)b * NN * HH + h;
  __shared__ float sr[256];
  int tid = threadIdx.x;

  float m = -3.0e38f;
  for (int n = tid; n < NN; n += 256)
    m = fmaxf(m, in[base + (size_t)n * HH]);
  sr[tid] = m; __syncthreads();
  for (int o = 128; o > 0; o >>= 1) {
    if (tid < o) sr[tid] = fmaxf(sr[tid], sr[tid + o]);
    __syncthreads();
  }
  float gm = sr[0]; __syncthreads();

  float s = 0.f;
  for (int n = tid; n < NN; n += 256)
    s += __expf(in[base + (size_t)n * HH] - gm);
  sr[tid] = s; __syncthreads();
  for (int o = 128; o > 0; o >>= 1) {
    if (tid < o) sr[tid] += sr[tid + o];
    __syncthreads();
  }
  float inv = 1.0f / sr[0];

  for (int n = tid; n < NN; n += 256)
    out[base + (size_t)n * HH] = __expf(in[base + (size_t)n * HH] - gm) * inv;
}

// ---------------------------------------------------------------------------
// K5: attn scale, in place on the bf16 v buffer: v *= aw[token, h]
// ---------------------------------------------------------------------------
__global__ void scale_v_kernel(const float* __restrict__ aw,
                               unsigned short* __restrict__ v, int n8) {
  int i = blockIdx.x * 256 + threadIdx.x;
  if (i >= n8) return;
  size_t e = (size_t)i * 8;
  int token = (int)(e >> 10);
  int c     = (int)(e & (CC - 1));
  int head  = c >> 5;
  float a = aw[(size_t)token * HH + head];
  Pack8 u;
  u.q = reinterpret_cast<uint4*>(v)[i];
#pragma unroll
  for (int t = 0; t < 8; t++) u.h[t] = f2bf(a * bf2f(u.h[t]));
  reinterpret_cast<uint4*>(v)[i] = u.q;
}

// ---------------------------------------------------------------------------
// K6: output projection (aw*v) @ Wo + bo -> fp32 d_out
// ---------------------------------------------------------------------------
__global__ __launch_bounds__(256) void out_gemm_kernel(
    const unsigned short* __restrict__ av,
    const unsigned short* __restrict__ pWo,
    const float* __restrict__ bo, float* __restrict__ out) {
  __shared__ unsigned short smA[16 * CC];     // 32 KB panel
  const int tid = threadIdx.x;
  const int mb  = blockIdx.x;
  {
    const uint4* src = reinterpret_cast<const uint4*>(av + (size_t)mb * 16 * CC);
    uint4* dst = reinterpret_cast<uint4*>(smA);
#pragma unroll
    for (int i = 0; i < 8; i++) dst[tid + i * 256] = src[tid + i * 256];
  }
  __syncthreads();

  const int wave = tid >> 5, lane = tid & 31, half = lane >> 4, mrow = lane & 15;
  const uint4* w4 = reinterpret_cast<const uint4*>(pWo);

  for (int t = wave; t < NT; t += 8) {
    v8f acc;
#pragma unroll
    for (int i = 0; i < 8; i++) acc[i] = 0.f;
    for (int kb = 0; kb < KT; kb++) {
      Frag16 A, Bw;
      int abase = mrow * CC + kb * 32 + half * 16;
      A.q[0] = *reinterpret_cast<const uint4*>(&smA[abase]);
      A.q[1] = *reinterpret_cast<const uint4*>(&smA[abase + 8]);
      size_t wb = ((size_t)(t * KT + kb) * 32 + lane) * 2;
      Bw.q[0] = w4[wb]; Bw.q[1] = w4[wb + 1];
      __builtin_prefetch((const void*)(w4 + wb + 64), 0, 1);
      acc = __builtin_amdgcn_wmma_f32_16x16x32_bf16(false, A.v, false, Bw.v,
                                                    (short)0, acc, false, false);
    }
    int   col = t * 16 + mrow;
    float bos = bo[col];
#pragma unroll
    for (int vi = 0; vi < 8; vi++) {
      int row = mb * 16 + vi + half * 8;
      out[(size_t)row * CC + col] = acc[vi] + bos;
    }
  }
}

// ---------------------------------------------------------------------------
// Host launcher
// ---------------------------------------------------------------------------
extern "C" void kernel_launch(void* const* d_in, const int* in_sizes, int n_in,
                              void* d_out, int out_size, void* d_ws, size_t ws_size,
                              hipStream_t stream) {
  (void)in_sizes; (void)n_in; (void)out_size; (void)ws_size;

  const float* x    = (const float*)d_in[0];
  const int*   mask = (const int*)d_in[1];
  const float* Wq   = (const float*)d_in[2];
  const float* bq   = (const float*)d_in[3];
  const float* Wk   = (const float*)d_in[4];
  const float* bk   = (const float*)d_in[5];
  const float* Wv   = (const float*)d_in[6];
  const float* bv   = (const float*)d_in[7];
  const float* W1   = (const float*)d_in[8];
  const float* b1   = (const float*)d_in[9];
  const float* W2   = (const float*)d_in[10];
  const float* b2   = (const float*)d_in[11];
  const float* Wo   = (const float*)d_in[12];
  const float* bo   = (const float*)d_in[13];
  float* out        = (float*)d_out;

  // Workspace layout (bytes)
  unsigned char* ws = (unsigned char*)d_ws;
  const size_t SZ_XB  = (size_t)MTOK * CC * 2;   // 64 MB  x bf16
  const size_t SZ_PW  = (size_t)CC * CC * 2;     // 2 MB   per packed weight
  const size_t SZ_VB  = (size_t)MTOK * CC * 2;   // 64 MB  v / aw*v bf16 (in place)
  const size_t SZ_AW  = (size_t)MTOK * HH * 4;   // 4 MB   aw buffers

  unsigned short* xb   = (unsigned short*)(ws);
  unsigned short* pWq  = (unsigned short*)(ws + SZ_XB);
  unsigned short* pWk  = (unsigned short*)(ws + SZ_XB + SZ_PW);
  unsigned short* pWv  = (unsigned short*)(ws + SZ_XB + 2 * SZ_PW);
  unsigned short* pWo  = (unsigned short*)(ws + SZ_XB + 3 * SZ_PW);
  unsigned short* vb   = (unsigned short*)(ws + SZ_XB + 4 * SZ_PW);
  float* aw_raw        = (float*)(ws + SZ_XB + 4 * SZ_PW + SZ_VB);
  float* aw_mlp        = (float*)(ws + SZ_XB + 4 * SZ_PW + SZ_VB + SZ_AW);
  float* aw_soft       = (float*)(ws + SZ_XB + 4 * SZ_PW + SZ_VB + 2 * SZ_AW);

  const int n8 = (MTOK * CC) / 8;                // 4,194,304

  // K0: x -> bf16
  cvt_bf16_kernel<<<(n8 + 255) / 256, 256, 0, stream>>>(x, xb, n8);

  // K1: pack weights into WMMA B-fragment layout
  pack_w_kernel<<<256, 256, 0, stream>>>(Wq, pWq);
  pack_w_kernel<<<256, 256, 0, stream>>>(Wk, pWk);
  pack_w_kernel<<<256, 256, 0, stream>>>(Wv, pWv);
  pack_w_kernel<<<256, 256, 0, stream>>>(Wo, pWo);

  // K2: fused QKV GEMMs (WMMA) + Hadamard dot
  qkv_dot_kernel<<<MTOK / 16, 256, 0, stream>>>(xb, pWq, pWk, pWv,
                                                bq, bk, bv, vb, aw_raw);

  // K3: dak MLP + mask
  dak_mlp_kernel<<<MTOK / 256, 256, 0, stream>>>(aw_raw, mask, W1, b1, W2, b2,
                                                 aw_mlp);

  // K4: softmax over N per (b, h)
  softmax_n_kernel<<<BB * HH, 256, 0, stream>>>(aw_mlp, aw_soft);

  // K5: v *= aw (in place, bf16)
  scale_v_kernel<<<(n8 + 255) / 256, 256, 0, stream>>>(aw_soft, vb, n8);

  // K6: (aw*v) @ Wo + bo -> out (WMMA)
  out_gemm_kernel<<<MTOK / 16, 256, 0, stream>>>(vb, pWo, bo, out);
}
